// GPT2Block_31842887533367
// MI455X (gfx1250) — matrix operand order repaired and checked by reading, not compiled, not run on hardware
//
#include <hip/hip_runtime.h>
#include <hip/hip_bf16.h>

typedef __attribute__((ext_vector_type(16))) _Float16 v16h;
typedef __attribute__((ext_vector_type(8)))  float    v8f;

#define BB 4
#define SS 2048
#define EE 4096
#define NTOK (BB * SS)          // 8192
#define SPLITC 256
#define QKVC 768
#define INNERC 1024
#define HDIM 128

static __device__ __forceinline__ v16h frag2(const _Float16* p0, const _Float16* p1) {
  union { uint4 u[2]; v16h v; } x;
  x.u[0] = *reinterpret_cast<const uint4*>(p0);
  x.u[1] = *reinterpret_cast<const uint4*>(p1);
  return x.v;
}

static __device__ __forceinline__ v8f wmma16(v16h a, v16h b, v8f c) {
  return __builtin_amdgcn_wmma_f32_16x16x32_f16(false, a, false, b, (short)0, c, false, false);
}

// ---------------- fp32 -> f16 convert ----------------
__global__ void cvt_f16(const float* __restrict__ s, _Float16* __restrict__ d, int n) {
  int i = blockIdx.x * blockDim.x + threadIdx.x;
  int stride = gridDim.x * blockDim.x;
  for (; i < n; i += stride) d[i] = (_Float16)s[i];
}

// ---------------- LayerNorm (fp32 in -> f16 out), one block per token ----------------
__global__ __launch_bounds__(256) void layernorm_f16(
    const float* __restrict__ X, const float* __restrict__ g,
    const float* __restrict__ b, _Float16* __restrict__ Y) {
  __shared__ float red[256];
  const int t = threadIdx.x;
  const size_t row = blockIdx.x;
  const float* x = X + row * EE;
  float vals[16];
  float s = 0.0f;
#pragma unroll
  for (int i = 0; i < 16; ++i) { vals[i] = x[t + i * 256]; s += vals[i]; }
  red[t] = s; __syncthreads();
  for (int off = 128; off > 0; off >>= 1) {
    if (t < off) red[t] += red[t + off];
    __syncthreads();
  }
  const float mu = red[0] * (1.0f / EE);
  __syncthreads();
  float vs = 0.0f;
#pragma unroll
  for (int i = 0; i < 16; ++i) { float d = vals[i] - mu; vs += d * d; }
  red[t] = vs; __syncthreads();
  for (int off = 128; off > 0; off >>= 1) {
    if (t < off) red[t] += red[t + off];
    __syncthreads();
  }
  const float rstd = rsqrtf(red[0] * (1.0f / EE) + 1e-5f);
#pragma unroll
  for (int i = 0; i < 16; ++i) {
    int c = t + i * 256;
    Y[row * EE + c] = (_Float16)((vals[i] - mu) * rstd * g[c] + b[c]);
  }
}

// ---------------- WMMA GEMM: C[M,N] = A[M,K]*B[K,N] + bias, 128x128 block tile -----------
// 8 waves as 4(M)x2(N); each wave owns 32x64 (2x4 accum tiles) -> 8 WMMA per K=32 step.
// MODE 0: f16 out          MODE 1: f16 out with exact GELU
// MODE 2: f32 out = v + Res[i]   MODE 3: f32 out += v (in place)
template <int MODE>
__global__ __launch_bounds__(256) void gemm_wmma(
    const _Float16* __restrict__ A, const _Float16* __restrict__ B,
    const float* __restrict__ bias, void* __restrict__ Cout,
    const float* __restrict__ Res, int M, int N, int K) {
  __shared__ _Float16 As[128 * 40];   // [row][k], pitch 40 (80B, 16B-aligned)
  __shared__ _Float16 Bst[128 * 40];  // [n][k],   pitch 40
  const int t = threadIdx.x;
  const int lane = t & 31, w = t >> 5;
  const int wm = w & 3, wn = w >> 2;
  const int hl = lane >> 4, ln = lane & 15;
  const int mBlock = blockIdx.y * 128, nBlock = blockIdx.x * 128;

  const int aRow = t >> 1, aCol = (t & 1) * 16;  // 128 rows x 32 k, 16 f16/thread
  const int bRow = t >> 3, bCol = (t & 7) * 16;  // 32 k x 128 n,  16 f16/thread

  v8f acc[2][4];
#pragma unroll
  for (int i = 0; i < 2; ++i)
#pragma unroll
    for (int j = 0; j < 4; ++j) acc[i][j] = (v8f){};

  for (int kb = 0; kb < K; kb += 32) {
    // A tile -> LDS (row-major)
    {
      const _Float16* src = A + (size_t)(mBlock + aRow) * K + kb + aCol;
      *reinterpret_cast<uint4*>(&As[aRow * 40 + aCol])     = *reinterpret_cast<const uint4*>(src);
      *reinterpret_cast<uint4*>(&As[aRow * 40 + aCol + 8]) = *reinterpret_cast<const uint4*>(src + 8);
    }
    // B tile -> LDS transposed ([n][k])
    {
      const _Float16* src = B + (size_t)(kb + bRow) * N + nBlock + bCol;
      uint4 b0 = *reinterpret_cast<const uint4*>(src);
      uint4 b1 = *reinterpret_cast<const uint4*>(src + 8);
      const _Float16* h0 = reinterpret_cast<const _Float16*>(&b0);
      const _Float16* h1 = reinterpret_cast<const _Float16*>(&b1);
#pragma unroll
      for (int j = 0; j < 8; ++j) {
        Bst[(bCol + j) * 40 + bRow]     = h0[j];
        Bst[(bCol + 8 + j) * 40 + bRow] = h1[j];
      }
    }
    __syncthreads();

    const _Float16* ar0 = &As[(wm * 32 + ln) * 40];
    const _Float16* ar1 = &As[(wm * 32 + 16 + ln) * 40];
    v16h a0 = frag2(ar0 + 16 * hl, ar0 + 16 * hl + 8);
    v16h a1 = frag2(ar1 + 16 * hl, ar1 + 16 * hl + 8);
#pragma unroll
    for (int j = 0; j < 4; ++j) {
      const _Float16* br = &Bst[(wn * 64 + j * 16 + ln) * 40];
      v16h bf = frag2(br + 16 * hl, br + 16 * hl + 8);
      acc[0][j] = wmma16(a0, bf, acc[0][j]);
      acc[1][j] = wmma16(a1, bf, acc[1][j]);
    }
    __syncthreads();
  }

#pragma unroll
  for (int i = 0; i < 2; ++i) {
#pragma unroll
    for (int j = 0; j < 4; ++j) {
      const int n = nBlock + wn * 64 + j * 16 + ln;
      const float bv = bias[n];
#pragma unroll
      for (int r = 0; r < 8; ++r) {
        const int m = mBlock + wm * 32 + i * 16 + 8 * hl + r;
        const size_t idx = (size_t)m * N + n;
        float v = acc[i][j][r] + bv;
        if (MODE == 0) {
          ((_Float16*)Cout)[idx] = (_Float16)v;
        } else if (MODE == 1) {
          v = 0.5f * v * (1.0f + erff(v * 0.70710678118654752f));
          ((_Float16*)Cout)[idx] = (_Float16)v;
        } else if (MODE == 2) {
          ((float*)Cout)[idx] = v + Res[idx];
        } else {
          ((float*)Cout)[idx] += v;
        }
      }
    }
  }
}

// ---------------- Causal flash attention: 64 queries per block, HD=128, 2 heads ----------
__global__ __launch_bounds__(256) void attn_flash(
    const _Float16* __restrict__ qkv, _Float16* __restrict__ outp) {
  constexpr int LDH = 136;  // pitch for 128-wide f16 rows (272B, 16B-aligned)
  constexpr int LDP = 72;   // pitch for 64-wide f16 rows  (144B, 16B-aligned)
  __shared__ _Float16 Qs[64 * LDH];
  __shared__ _Float16 KV[128 * LDP];   // K tile [key][d] pitch LDH (8704<=9216 f16), then V^T [d][key] pitch LDP
  __shared__ float    Sf[64 * 65];
  __shared__ _Float16 Ps[64 * LDP];
  __shared__ float m_run[64], l_run[64], fac_s[64];

  const int t = threadIdx.x;
  const int lane = t & 31, w = t >> 5;
  const int wm = w & 3, wn = w >> 2;
  const int hl = lane >> 4, ln = lane & 15;
  const int qTile = blockIdx.x;
  const int b = blockIdx.y >> 1, h = blockIdx.y & 1;
  const size_t rowBase = (size_t)b * SS;
  const int qBase = qTile * 64;
  const int qOff = h * HDIM, kOff = SPLITC + h * HDIM, vOff = 2 * SPLITC + h * HDIM;

  // load Q tile (64 x 128)
#pragma unroll
  for (int i = 0; i < 4; ++i) {
    int o = t + i * 256;
    int r = o >> 4, c = (o & 15) * 8;
    uint4 v = *reinterpret_cast<const uint4*>(qkv + (rowBase + qBase + r) * QKVC + qOff + c);
    *reinterpret_cast<uint4*>(&Qs[r * LDH + c]) = v;
  }
  if (t < 64) { m_run[t] = -3.0e38f; l_run[t] = 0.0f; }

  v8f o0 = {}, o1 = {}, o2 = {}, o3 = {};

  for (int kt = 0; kt <= qTile; ++kt) {
    const int kBase = kt * 64;
    __syncthreads();  // protect KV/Ps from previous iteration's consumers
    // load K tile (row-major [key][d]) into KV
#pragma unroll
    for (int i = 0; i < 4; ++i) {
      int o = t + i * 256;
      int r = o >> 4, c = (o & 15) * 8;
      uint4 v = *reinterpret_cast<const uint4*>(qkv + (rowBase + kBase + r) * QKVC + kOff + c);
      *reinterpret_cast<uint4*>(&KV[r * LDH + c]) = v;
    }
    __syncthreads();

    // S = Q K^T : each wave 16 q-rows x 32 keys
    v8f s0 = {}, s1 = {};
    {
      const _Float16* qrow = &Qs[(wm * 16 + ln) * LDH];
      const _Float16* kr0 = &KV[(wn * 32 + ln) * LDH];
      const _Float16* kr1 = &KV[(wn * 32 + 16 + ln) * LDH];
#pragma unroll
      for (int d = 0; d < HDIM; d += 32) {
        v16h af = frag2(qrow + d + 16 * hl, qrow + d + 16 * hl + 8);
        v16h bf0 = frag2(kr0 + d + 16 * hl, kr0 + d + 16 * hl + 8);
        v16h bf1 = frag2(kr1 + d + 16 * hl, kr1 + d + 16 * hl + 8);
        s0 = wmma16(af, bf0, s0);
        s1 = wmma16(af, bf1, s1);
      }
      const int m = wm * 16 + 8 * hl;
      const int n0 = wn * 32 + ln, n1 = n0 + 16;
#pragma unroll
      for (int r = 0; r < 8; ++r) {
        Sf[(m + r) * 65 + n0] = s0[r];
        Sf[(m + r) * 65 + n1] = s1[r];
      }
    }
    __syncthreads();

    // load V tile transposed into KV ([d][key], pitch LDP) — K no longer needed
#pragma unroll
    for (int i = 0; i < 4; ++i) {
      int o = t + i * 256;
      int r = o >> 4, c = (o & 15) * 8;
      uint4 v = *reinterpret_cast<const uint4*>(qkv + (rowBase + kBase + r) * QKVC + vOff + c);
      const _Float16* vh = reinterpret_cast<const _Float16*>(&v);
#pragma unroll
      for (int j = 0; j < 8; ++j) KV[(c + j) * LDP + r] = vh[j];
    }
    // online softmax per query row (threads 0..63)
    if (t < 64) {
      const int qg = qBase + t;
      const float mOld = m_run[t];
      float mx = mOld;
      for (int j = 0; j < 64; ++j) {
        float v = Sf[t * 65 + j] * 0.08838834764831845f;  // 1/sqrt(128)
        if (kBase + j > qg) v = -10000.0f;                 // reference MASKED_BIAS
        Sf[t * 65 + j] = v;
        mx = fmaxf(mx, v);
      }
      const float fac = __expf(mOld - mx);
      float l = l_run[t] * fac;
      for (int j = 0; j < 64; ++j) {
        float e = __expf(Sf[t * 65 + j] - mx);
        Ps[t * LDP + j] = (_Float16)e;
        l += e;
      }
      m_run[t] = mx; l_run[t] = l; fac_s[t] = fac;
    }
    __syncthreads();

    // rescale accumulators, then O += P @ V
    float fr[8];
#pragma unroll
    for (int r = 0; r < 8; ++r) fr[r] = fac_s[wm * 16 + 8 * hl + r];
#pragma unroll
    for (int r = 0; r < 8; ++r) { o0[r] *= fr[r]; o1[r] *= fr[r]; o2[r] *= fr[r]; o3[r] *= fr[r]; }

    const _Float16* prow = &Ps[(wm * 16 + ln) * LDP];
    const _Float16* vr0 = &KV[(wn * 64 + ln) * LDP];
    const _Float16* vr1 = &KV[(wn * 64 + 16 + ln) * LDP];
    const _Float16* vr2 = &KV[(wn * 64 + 32 + ln) * LDP];
    const _Float16* vr3 = &KV[(wn * 64 + 48 + ln) * LDP];
#pragma unroll
    for (int ks = 0; ks < 64; ks += 32) {
      v16h af = frag2(prow + ks + 16 * hl, prow + ks + 16 * hl + 8);
      v16h b0 = frag2(vr0 + ks + 16 * hl, vr0 + ks + 16 * hl + 8);
      v16h b1 = frag2(vr1 + ks + 16 * hl, vr1 + ks + 16 * hl + 8);
      v16h b2 = frag2(vr2 + ks + 16 * hl, vr2 + ks + 16 * hl + 8);
      v16h b3 = frag2(vr3 + ks + 16 * hl, vr3 + ks + 16 * hl + 8);
      o0 = wmma16(af, b0, o0);
      o1 = wmma16(af, b1, o1);
      o2 = wmma16(af, b2, o2);
      o3 = wmma16(af, b3, o3);
    }
  }
  __syncthreads();

  // normalize by l and write merged-head output [tok][h*128 + d] as f16
  const int m = wm * 16 + 8 * hl;
#pragma unroll
  for (int r = 0; r < 8; ++r) {
    const float inv = 1.0f / l_run[m + r];
    const size_t orow = (rowBase + qBase + m + r) * SPLITC + h * HDIM + wn * 64;
    outp[orow + ln]      = (_Float16)(o0[r] * inv);
    outp[orow + 16 + ln] = (_Float16)(o1[r] * inv);
    outp[orow + 32 + ln] = (_Float16)(o2[r] * inv);
    outp[orow + 48 + ln] = (_Float16)(o3[r] * inv);
  }
}

extern "C" void kernel_launch(void* const* d_in, const int* in_sizes, int n_in,
                              void* d_out, int out_size, void* d_ws, size_t ws_size,
                              hipStream_t stream) {
  (void)in_sizes; (void)n_in; (void)out_size; (void)ws_size;
  const float* hs          = (const float*)d_in[0];
  const float* ln1_g       = (const float*)d_in[1];
  const float* ln1_b       = (const float*)d_in[2];
  const float* c_attn_w    = (const float*)d_in[3];
  const float* c_attn_b    = (const float*)d_in[4];
  const float* attn_proj_w = (const float*)d_in[5];
  const float* attn_proj_b = (const float*)d_in[6];
  const float* ln2_g       = (const float*)d_in[7];
  const float* ln2_b       = (const float*)d_in[8];
  const float* fc_w        = (const float*)d_in[9];
  const float* fc_b        = (const float*)d_in[10];
  const float* mlp_proj_w  = (const float*)d_in[11];
  const float* mlp_proj_b  = (const float*)d_in[12];
  float* out = (float*)d_out;

  char* ws = (char*)d_ws;
  _Float16* W1   = (_Float16*)(ws + 0);           // 4096*768
  _Float16* W2   = (_Float16*)(ws + 6291456);     // 256*4096
  _Float16* W3   = (_Float16*)(ws + 8388608);     // 4096*1024
  _Float16* W4   = (_Float16*)(ws + 16777216);    // 1024*4096
  _Float16* X16  = (_Float16*)(ws + 25165824);    // 8192*4096 (ln1 then ln2 output)
  _Float16* QKV  = (_Float16*)(ws + 92274688);    // 8192*768
  _Float16* ATT  = (_Float16*)(ws + 104857600);   // 8192*256
  _Float16* H16  = (_Float16*)(ws + 109051904);   // 8192*1024

  // weights fp32 -> f16
  cvt_f16<<<4096, 256, 0, stream>>>(c_attn_w, W1, EE * QKVC);
  cvt_f16<<<4096, 256, 0, stream>>>(attn_proj_w, W2, SPLITC * EE);
  cvt_f16<<<4096, 256, 0, stream>>>(fc_w, W3, EE * INNERC);
  cvt_f16<<<4096, 256, 0, stream>>>(mlp_proj_w, W4, INNERC * EE);

  // LN1
  layernorm_f16<<<NTOK, 256, 0, stream>>>(hs, ln1_g, ln1_b, X16);
  // QKV = LN1 @ c_attn_w + b   [8192 x 768]
  gemm_wmma<0><<<dim3(QKVC / 128, NTOK / 128), 256, 0, stream>>>(
      X16, W1, c_attn_b, (void*)QKV, nullptr, NTOK, QKVC, EE);
  // causal attention per (b, head), 64-query tiles
  attn_flash<<<dim3(SS / 64, BB * 2), 256, 0, stream>>>(QKV, ATT);
  // hidden = ATT @ attn_proj_w + b + residual  (fp32 into d_out)
  gemm_wmma<2><<<dim3(EE / 128, NTOK / 128), 256, 0, stream>>>(
      ATT, W2, attn_proj_b, (void*)out, hs, NTOK, EE, SPLITC);
  // LN2
  layernorm_f16<<<NTOK, 256, 0, stream>>>(out, ln2_g, ln2_b, X16);
  // H = gelu(LN2 @ fc_w + b)  [8192 x 1024]
  gemm_wmma<1><<<dim3(INNERC / 128, NTOK / 128), 256, 0, stream>>>(
      X16, W3, fc_b, (void*)H16, nullptr, NTOK, INNERC, EE);
  // out += H @ mlp_proj_w + b  (fp32 accumulate in place)
  gemm_wmma<3><<<dim3(EE / 128, NTOK / 128), 256, 0, stream>>>(
      H16, W4, mlp_proj_b, (void*)out, nullptr, NTOK, EE, INNERC);
}
